// RNNDAModel_88313117540681
// MI455X (gfx1250) — matrix operand order repaired and checked by reading, not compiled, not run on hardware
//
#include <hip/hip_runtime.h>
#include <hip/hip_bf16.h>

// ---------------- problem constants ----------------
#define BB 4096
#define WW 50
#define FF 64
#define HH 128
#define GG 512   // 4*H

typedef __attribute__((ext_vector_type(16))) _Float16 v16h;
typedef __attribute__((ext_vector_type(8)))  _Float16 v8h;
typedef __attribute__((ext_vector_type(8)))  float    v8f;

__device__ __forceinline__ float sigm(float x) {
    return 1.0f / (1.0f + __expf(-x));
}
__device__ __forceinline__ float mytanh(float x) {
    x = fminf(15.0f, fmaxf(-15.0f, x));
    float e = __expf(2.0f * x);
    return (e - 1.0f) / (e + 1.0f);
}

// A-fragment (16x32 f16, row-major tile in LDS, rowStride halves, 16B-aligned).
// Layout: lanes 0-15 hold row M=lane, K = {0..15};
//         lanes 16-31 hold row M=lane-16, K = {16..31}.
__device__ __forceinline__ v16h load_a_frag(const _Float16* base, int rowStride) {
    int lane = threadIdx.x & 31;
    const _Float16* p = base + (lane & 15) * rowStride + ((lane >> 4) << 4);
    v8h lo = *(const v8h*)(p);
    v8h hi = *(const v8h*)(p + 8);
    v16h r;
#pragma unroll
    for (int i = 0; i < 8; ++i) { r[i] = lo[i]; r[8 + i] = hi[i]; }
    return r;
}

// B-fragment pre-packed in global memory: [(kt*NT + nt)*32 + lane] * 16 halves.
__device__ __forceinline__ v16h load_b_frag(const _Float16* frag, int kt, int nt, int NT) {
    int lane = threadIdx.x & 31;
    const _Float16* p = frag + ((size_t)((kt * NT + nt) * 32 + lane) << 4);
    return *(const v16h*)p;
}

// ---------------- weight fragment packing ----------------
// Packs W (Wout x Win, out-major) transposed GEMM operand into WMMA B-frag
// layout.  Win = colsA + colsB (optional concatenated second source).
// Frag element ((kt*NT+nt)*32+lane)*16+i  <-  W[o][k],
//   o = nt*16 + (lane&15),  k = kt*32 + (lane>>4)*16 + i.
__global__ void pack_frags_kernel(const float* __restrict__ A, int lda, int colsA,
                                  const float* __restrict__ Bsrc, int ldb,
                                  int Win, int Wout, int NT,
                                  _Float16* __restrict__ dst, int total) {
    int idx = blockIdx.x * blockDim.x + threadIdx.x;
    if (idx >= total) return;
    int i    = idx & 15;
    int lane = (idx >> 4) & 31;
    int nt   = (idx >> 9) % NT;
    int kt   = (idx >> 9) / NT;
    int o = nt * 16 + (lane & 15);
    int k = kt * 32 + ((lane >> 4) << 4) + i;
    float v = 0.0f;
    if (o < Wout && k < Win) {
        if (k < colsA) v = A[(size_t)o * lda + k];
        else           v = Bsrc[(size_t)o * ldb + (k - colsA)];
    }
    dst[idx] = (_Float16)v;
}

// ---------------- zero init of LSTM states ----------------
__global__ void zero_state_kernel(float* h, float* s, float* d, float* c, int n) {
    int i = blockIdx.x * blockDim.x + threadIdx.x;
    if (i < n) { h[i] = 0.0f; s[i] = 0.0f; d[i] = 0.0f; c[i] = 0.0f; }
}

// ---------------- Xpart: time-invariant input-attention term ----------------
// Xpart[b][k][j] = enc_ab1[j] + sum_w enc_aW1[j][256+w] * seq[b][w][k]
__global__ void __launch_bounds__(64) xpart_kernel(const float* __restrict__ seq,
                                                   const float* __restrict__ aW1,
                                                   const float* __restrict__ ab1,
                                                   _Float16* __restrict__ xpart) {
    __shared__ float sv[WW];
    int bk = blockIdx.x;            // b*64 + k
    int b = bk >> 6, k = bk & 63;
    int j = threadIdx.x;
    if (j < WW) sv[j] = seq[((size_t)b * WW + j) * FF + k];
    __syncthreads();
    float acc = ab1[j];
    const float* wr = aW1 + (size_t)j * 306 + 256;
#pragma unroll 10
    for (int w = 0; w < WW; ++w) acc += wr[w] * sv[w];
    xpart[(size_t)bk * 64 + j] = (_Float16)acc;
}

// ---------------- encoder input attention (per step) ----------------
__global__ void __launch_bounds__(128) enc_attn_kernel(
    const float* __restrict__ h, const float* __restrict__ s,
    const _Float16* __restrict__ a1frag, const _Float16* __restrict__ xpart,
    const float* __restrict__ seq, const float* __restrict__ aW2,
    const float* __restrict__ ab2, _Float16* __restrict__ xt16, int t) {
    __shared__ __attribute__((aligned(32))) _Float16 As[16 * 256];
    __shared__ float Ps[16 * 64];
    __shared__ float Es[16 * 64];
    __shared__ float w2s[64];
    int tid = threadIdx.x;
    int b0 = blockIdx.x * 16;
    if (tid < 64) w2s[tid] = aW2[tid];
    for (int e = tid; e < 16 * 256; e += 128) {
        int m = e >> 8, k = e & 255;
        int b = b0 + m;
        float v = (k < HH) ? h[(size_t)b * HH + k] : s[(size_t)b * HH + (k - HH)];
        As[e] = (_Float16)v;
    }
    __syncthreads();
    int wave = tid >> 5, lane = tid & 31;
    {   // hsproj = [h|s] @ W1hs.T  : K=256 (8 steps), N=64 (4 tiles, 1/wave)
        v8f acc = {};
#pragma unroll
        for (int kt = 0; kt < 8; ++kt) {
            v16h a  = load_a_frag(As + kt * 32, 256);
            v16h bf = load_b_frag(a1frag, kt, wave, 4);
            acc = __builtin_amdgcn_wmma_f32_16x16x32_f16(false, a, false, bf,
                                                         (short)0, acc, false, false);
        }
        int n0 = wave * 16 + (lane & 15);
        int m0 = (lane >> 4) << 3;
#pragma unroll
        for (int v = 0; v < 8; ++v) Ps[(m0 + v) * 64 + n0] = acc[v];
    }
    __syncthreads();
    float c2 = ab2[0];
    for (int e = tid; e < 16 * 64; e += 128) {
        int m = e >> 6, k = e & 63;
        const _Float16* xp = xpart + ((size_t)(b0 + m) * 64 + k) * 64;
        float acc = c2;
#pragma unroll 8
        for (int j = 0; j < 64; ++j)
            acc += w2s[j] * mytanh(Ps[m * 64 + j] + (float)xp[j]);
        Es[e] = acc;
    }
    __syncthreads();
    if (tid < 16) {   // softmax over F=64
        int m = tid;
        float mx = -1e30f;
        for (int k = 0; k < 64; ++k) mx = fmaxf(mx, Es[m * 64 + k]);
        float sum = 0.0f;
        for (int k = 0; k < 64; ++k) { float ev = __expf(Es[m * 64 + k] - mx); Es[m * 64 + k] = ev; sum += ev; }
        float inv = 1.0f / sum;
        for (int k = 0; k < 64; ++k) Es[m * 64 + k] *= inv;
    }
    __syncthreads();
    for (int e = tid; e < 16 * 64; e += 128) {
        int m = e >> 6, k = e & 63;
        int b = b0 + m;
        xt16[(size_t)b * 64 + k] =
            (_Float16)(Es[e] * seq[((size_t)b * WW + t) * FF + k]);
    }
}

// ---------------- encoder LSTM step ----------------
__global__ void __launch_bounds__(128) enc_lstm_kernel(
    const _Float16* __restrict__ xt16, float* __restrict__ h, float* __restrict__ s,
    const _Float16* __restrict__ wfrag, const float* __restrict__ bih,
    const float* __restrict__ bhh, _Float16* __restrict__ henc, int t) {
    __shared__ __attribute__((aligned(32))) _Float16 As[16 * 192];
    __shared__ float Gs[16 * GG];
    int tid = threadIdx.x;
    int b0 = blockIdx.x * 16;
    for (int e = tid; e < 16 * 192; e += 128) {
        int m = e / 192, k = e - m * 192;
        int b = b0 + m;
        As[e] = (k < FF) ? xt16[(size_t)b * FF + k]
                         : (_Float16)h[(size_t)b * HH + (k - FF)];
    }
    __syncthreads();
    int wave = tid >> 5, lane = tid & 31;
    for (int nt = wave; nt < 32; nt += 4) {   // g = [x~|h] @ [Wih|Whh].T : K=192
        v8f acc = {};
#pragma unroll
        for (int kt = 0; kt < 6; ++kt) {
            v16h a  = load_a_frag(As + kt * 32, 192);
            v16h bf = load_b_frag(wfrag, kt, nt, 32);
            acc = __builtin_amdgcn_wmma_f32_16x16x32_f16(false, a, false, bf,
                                                         (short)0, acc, false, false);
        }
        int n0 = nt * 16 + (lane & 15);
        int m0 = (lane >> 4) << 3;
#pragma unroll
        for (int v = 0; v < 8; ++v) Gs[(m0 + v) * GG + n0] = acc[v];
    }
    __syncthreads();
    for (int e = tid; e < 16 * HH; e += 128) {
        int m = e >> 7, j = e & 127;
        int b = b0 + m;
        float gi = Gs[m * GG + j]           + bih[j]           + bhh[j];
        float gf = Gs[m * GG + 128 + j]     + bih[128 + j]     + bhh[128 + j];
        float gg = Gs[m * GG + 256 + j]     + bih[256 + j]     + bhh[256 + j];
        float go = Gs[m * GG + 384 + j]     + bih[384 + j]     + bhh[384 + j];
        float cin = s[(size_t)b * HH + j];
        float cn = sigm(gf) * cin + sigm(gi) * mytanh(gg);
        float hn = sigm(go) * mytanh(cn);
        s[(size_t)b * HH + j] = cn;
        h[(size_t)b * HH + j] = hn;
        henc[((size_t)b * WW + t) * HH + j] = (_Float16)hn;
    }
}

// ---------------- HencProj (one-time): Henc @ dec_aW1[:,256:384].T + dec_ab1 ----
__global__ void __launch_bounds__(128) hencproj_kernel(
    const _Float16* __restrict__ henc, const _Float16* __restrict__ wfrag,
    const float* __restrict__ ab1, _Float16* __restrict__ hproj) {
    __shared__ __attribute__((aligned(32))) _Float16 As[16 * 128];
    int tid = threadIdx.x;
    int r0 = blockIdx.x * 16;    // flattened row b*W + w
    for (int e = tid; e < 16 * 128; e += 128)
        As[e] = henc[(size_t)(r0 + (e >> 7)) * HH + (e & 127)];
    __syncthreads();
    int wave = tid >> 5, lane = tid & 31;
    v8f acc = {};
#pragma unroll
    for (int kt = 0; kt < 4; ++kt) {
        v16h a  = load_a_frag(As + kt * 32, 128);
        v16h bf = load_b_frag(wfrag, kt, wave, 4);
        acc = __builtin_amdgcn_wmma_f32_16x16x32_f16(false, a, false, bf,
                                                     (short)0, acc, false, false);
    }
    int n0 = wave * 16 + (lane & 15);
    int m0 = (lane >> 4) << 3;
    float bv = ab1[n0];
#pragma unroll
    for (int v = 0; v < 8; ++v)
        hproj[(size_t)(r0 + m0 + v) * 64 + n0] = (_Float16)(acc[v] + bv);
}

// ---------------- decoder temporal attention (per step) ----------------
__global__ void __launch_bounds__(128) dec_attn_kernel(
    const float* __restrict__ d, const float* __restrict__ c,
    const _Float16* __restrict__ a1frag, const _Float16* __restrict__ hproj,
    const _Float16* __restrict__ henc, const float* __restrict__ aW2,
    const float* __restrict__ ab2, const float* __restrict__ fcW,
    const float* __restrict__ fcb, const float* __restrict__ seq,
    float* __restrict__ ctx, float* __restrict__ ytil, int t) {
    __shared__ __attribute__((aligned(32))) _Float16 As[16 * 256];
    __shared__ float Ps[16 * 64];
    __shared__ float Es[16 * 64];
    __shared__ float Cs[16 * 128];
    __shared__ float w2s[64];
    int tid = threadIdx.x;
    int b0 = blockIdx.x * 16;
    if (tid < 64) w2s[tid] = aW2[tid];
    for (int e = tid; e < 16 * 256; e += 128) {
        int m = e >> 8, k = e & 255;
        int b = b0 + m;
        float v = (k < HH) ? d[(size_t)b * HH + k] : c[(size_t)b * HH + (k - HH)];
        As[e] = (_Float16)v;
    }
    __syncthreads();
    int wave = tid >> 5, lane = tid & 31;
    {   // dcproj = [d|c] @ W1dc.T
        v8f acc = {};
#pragma unroll
        for (int kt = 0; kt < 8; ++kt) {
            v16h a  = load_a_frag(As + kt * 32, 256);
            v16h bf = load_b_frag(a1frag, kt, wave, 4);
            acc = __builtin_amdgcn_wmma_f32_16x16x32_f16(false, a, false, bf,
                                                         (short)0, acc, false, false);
        }
        int n0 = wave * 16 + (lane & 15);
        int m0 = (lane >> 4) << 3;
#pragma unroll
        for (int v = 0; v < 8; ++v) Ps[(m0 + v) * 64 + n0] = acc[v];
    }
    __syncthreads();
    float c2 = ab2[0];
    for (int e = tid; e < 16 * WW; e += 128) {
        int m = e / WW, w = e - m * WW;
        const _Float16* hp = hproj + ((size_t)(b0 + m) * WW + w) * 64;
        float acc = c2;
#pragma unroll 8
        for (int j = 0; j < 64; ++j)
            acc += w2s[j] * mytanh(Ps[m * 64 + j] + (float)hp[j]);
        Es[m * 64 + w] = acc;
    }
    __syncthreads();
    if (tid < 16) {   // softmax over W=50
        int m = tid;
        float mx = -1e30f;
        for (int w = 0; w < WW; ++w) mx = fmaxf(mx, Es[m * 64 + w]);
        float sum = 0.0f;
        for (int w = 0; w < WW; ++w) { float ev = __expf(Es[m * 64 + w] - mx); Es[m * 64 + w] = ev; sum += ev; }
        float inv = 1.0f / sum;
        for (int w = 0; w < WW; ++w) Es[m * 64 + w] *= inv;
    }
    __syncthreads();
    for (int e = tid; e < 16 * HH; e += 128) {   // context = beta . Henc
        int m = e >> 7, j = e & 127;
        int b = b0 + m;
        const _Float16* hb = henc + ((size_t)b * WW) * HH + j;
        float acc = 0.0f;
#pragma unroll 10
        for (int w = 0; w < WW; ++w) acc += Es[m * 64 + w] * (float)hb[(size_t)w * HH];
        Cs[e] = acc;
        ctx[(size_t)b * HH + j] = acc;
    }
    __syncthreads();
    if (tid < 16) {   // y_tilde = fc_W . [context, y_t] + fc_b
        int m = tid, b = b0 + m;
        float acc = fcb[0] + fcW[128] * seq[((size_t)b * WW + t) * FF + 1];
        for (int j = 0; j < HH; ++j) acc += fcW[j] * Cs[m * 128 + j];
        ytil[b] = acc;
    }
}

// ---------------- decoder LSTM step ----------------
__global__ void __launch_bounds__(128) dec_lstm_kernel(
    const float* __restrict__ ytil, const float* __restrict__ wih,
    float* __restrict__ d, float* __restrict__ c,
    const _Float16* __restrict__ wfrag, const float* __restrict__ bih,
    const float* __restrict__ bhh) {
    __shared__ __attribute__((aligned(32))) _Float16 As[16 * 128];
    __shared__ float Gs[16 * GG];
    int tid = threadIdx.x;
    int b0 = blockIdx.x * 16;
    for (int e = tid; e < 16 * 128; e += 128)
        As[e] = (_Float16)d[(size_t)(b0 + (e >> 7)) * HH + (e & 127)];
    __syncthreads();
    int wave = tid >> 5, lane = tid & 31;
    for (int nt = wave; nt < 32; nt += 4) {   // d @ Whh.T : K=128
        v8f acc = {};
#pragma unroll
        for (int kt = 0; kt < 4; ++kt) {
            v16h a  = load_a_frag(As + kt * 32, 128);
            v16h bf = load_b_frag(wfrag, kt, nt, 32);
            acc = __builtin_amdgcn_wmma_f32_16x16x32_f16(false, a, false, bf,
                                                         (short)0, acc, false, false);
        }
        int n0 = nt * 16 + (lane & 15);
        int m0 = (lane >> 4) << 3;
#pragma unroll
        for (int v = 0; v < 8; ++v) Gs[(m0 + v) * GG + n0] = acc[v];
    }
    __syncthreads();
    for (int e = tid; e < 16 * HH; e += 128) {
        int m = e >> 7, j = e & 127;
        int b = b0 + m;
        float yt = ytil[b];
        float gi = Gs[m * GG + j]       + bih[j]       + bhh[j]       + yt * wih[j];
        float gf = Gs[m * GG + 128 + j] + bih[128 + j] + bhh[128 + j] + yt * wih[128 + j];
        float gg = Gs[m * GG + 256 + j] + bih[256 + j] + bhh[256 + j] + yt * wih[256 + j];
        float go = Gs[m * GG + 384 + j] + bih[384 + j] + bhh[384 + j] + yt * wih[384 + j];
        float cin = c[(size_t)b * HH + j];
        float cn = sigm(gf) * cin + sigm(gi) * mytanh(gg);
        float hn = sigm(go) * mytanh(cn);
        c[(size_t)b * HH + j] = cn;
        d[(size_t)b * HH + j] = hn;
    }
}

// ---------------- final projection ----------------
__global__ void final_kernel(const float* __restrict__ d, const float* __restrict__ ctx,
                             const float* __restrict__ ffW, const float* __restrict__ ffb,
                             float* __restrict__ out) {
    int b = blockIdx.x * blockDim.x + threadIdx.x;
    if (b >= BB) return;
    float acc = ffb[0];
#pragma unroll 8
    for (int j = 0; j < HH; ++j)
        acc += ffW[j] * d[(size_t)b * HH + j] + ffW[128 + j] * ctx[(size_t)b * HH + j];
    out[b] = acc;
}

// ---------------- host launcher ----------------
extern "C" void kernel_launch(void* const* d_in, const int* in_sizes, int n_in,
                              void* d_out, int out_size, void* d_ws, size_t ws_size,
                              hipStream_t stream) {
    (void)in_sizes; (void)n_in; (void)out_size; (void)ws_size;
    const float* seq      = (const float*)d_in[0];
    const float* enc_Wih  = (const float*)d_in[1];
    const float* enc_Whh  = (const float*)d_in[2];
    const float* enc_bih  = (const float*)d_in[3];
    const float* enc_bhh  = (const float*)d_in[4];
    const float* enc_aW1  = (const float*)d_in[5];
    const float* enc_ab1  = (const float*)d_in[6];
    const float* enc_aW2  = (const float*)d_in[7];
    const float* enc_ab2  = (const float*)d_in[8];
    const float* dec_Wih  = (const float*)d_in[9];
    const float* dec_Whh  = (const float*)d_in[10];
    const float* dec_bih  = (const float*)d_in[11];
    const float* dec_bhh  = (const float*)d_in[12];
    const float* dec_aW1  = (const float*)d_in[13];
    const float* dec_ab1  = (const float*)d_in[14];
    const float* dec_aW2  = (const float*)d_in[15];
    const float* dec_ab2  = (const float*)d_in[16];
    const float* fc_W     = (const float*)d_in[17];
    const float* fc_b     = (const float*)d_in[18];
    const float* ff_W     = (const float*)d_in[19];
    const float* ff_b     = (const float*)d_in[20];
    float* out = (float*)d_out;

    // ---- carve workspace ----
    char* p = (char*)d_ws;
    auto carve = [&](size_t bytes) -> char* {
        char* r = p;
        p += (bytes + 255) & ~(size_t)255;
        return r;
    };
    float*     h        = (float*)carve((size_t)BB * HH * 4);
    float*     s        = (float*)carve((size_t)BB * HH * 4);
    float*     dstate   = (float*)carve((size_t)BB * HH * 4);
    float*     cstate   = (float*)carve((size_t)BB * HH * 4);
    float*     ctx      = (float*)carve((size_t)BB * HH * 4);
    float*     ytil     = (float*)carve((size_t)BB * 4);
    _Float16*  xt16     = (_Float16*)carve((size_t)BB * FF * 2);
    _Float16*  henc16   = (_Float16*)carve((size_t)BB * WW * HH * 2);
    _Float16*  xpart16  = (_Float16*)carve((size_t)BB * FF * 64 * 2);
    _Float16*  hproj16  = (_Float16*)carve((size_t)BB * WW * 64 * 2);
    _Float16*  encWfrag = (_Float16*)carve((size_t)6 * 32 * 512 * 2);
    _Float16*  encA1f   = (_Float16*)carve((size_t)8 * 4 * 512 * 2);
    _Float16*  decA1f   = (_Float16*)carve((size_t)8 * 4 * 512 * 2);
    _Float16*  decHPf   = (_Float16*)carve((size_t)4 * 4 * 512 * 2);
    _Float16*  decWfrag = (_Float16*)carve((size_t)4 * 32 * 512 * 2);

    // ---- init + weight packing ----
    zero_state_kernel<<<(BB * HH + 255) / 256, 256, 0, stream>>>(h, s, dstate, cstate, BB * HH);

    int tot;
    tot = 6 * 32 * 512;   // enc LSTM  [Wih(512x64) | Whh(512x128)]  K=192,N=512
    pack_frags_kernel<<<(tot + 255) / 256, 256, 0, stream>>>(
        enc_Wih, 64, 64, enc_Whh, 128, 192, 512, 32, encWfrag, tot);
    tot = 8 * 4 * 512;    // enc attn  enc_aW1[:, :256]  K=256,N=64
    pack_frags_kernel<<<(tot + 255) / 256, 256, 0, stream>>>(
        enc_aW1, 306, 256, enc_aW1, 306, 256, 64, 4, encA1f, tot);
    tot = 8 * 4 * 512;    // dec attn  dec_aW1[:, :256]  K=256,N=64
    pack_frags_kernel<<<(tot + 255) / 256, 256, 0, stream>>>(
        dec_aW1, 384, 256, dec_aW1, 384, 256, 64, 4, decA1f, tot);
    tot = 4 * 4 * 512;    // HencProj  dec_aW1[:, 256:384]  K=128,N=64
    pack_frags_kernel<<<(tot + 255) / 256, 256, 0, stream>>>(
        dec_aW1 + 256, 384, 128, dec_aW1, 384, 128, 64, 4, decHPf, tot);
    tot = 4 * 32 * 512;   // dec LSTM  dec_Whh(512x128)  K=128,N=512
    pack_frags_kernel<<<(tot + 255) / 256, 256, 0, stream>>>(
        dec_Whh, 128, 128, dec_Whh, 128, 128, 512, 32, decWfrag, tot);

    // time-invariant input-attention term
    xpart_kernel<<<BB * FF, 64, 0, stream>>>(seq, enc_aW1, enc_ab1, xpart16);

    // ---- encoder ----
    for (int t = 0; t < WW; ++t) {
        enc_attn_kernel<<<BB / 16, 128, 0, stream>>>(
            h, s, encA1f, xpart16, seq, enc_aW2, enc_ab2, xt16, t);
        enc_lstm_kernel<<<BB / 16, 128, 0, stream>>>(
            xt16, h, s, encWfrag, enc_bih, enc_bhh, henc16, t);
    }

    // one-time Henc projection for decoder attention (folds dec_ab1)
    hencproj_kernel<<<(BB * WW) / 16, 128, 0, stream>>>(henc16, decHPf, dec_ab1, hproj16);

    // ---- decoder ----
    for (int t = 0; t < WW; ++t) {
        dec_attn_kernel<<<BB / 16, 128, 0, stream>>>(
            dstate, cstate, decA1f, hproj16, henc16, dec_aW2, dec_ab2,
            fc_W, fc_b, seq, ctx, ytil, t);
        dec_lstm_kernel<<<BB / 16, 128, 0, stream>>>(
            ytil, dec_Wih, dstate, cstate, decWfrag, dec_bih, dec_bhh);
    }

    // ---- final projection ----
    final_kernel<<<(BB + 255) / 256, 256, 0, stream>>>(dstate, ctx, ff_W, ff_b, out);
}
